// TSPPolicy_12756052869476
// MI455X (gfx1250) — matrix operand order repaired and checked by reading, not compiled, not run on hardware
//
#include <hip/hip_runtime.h>
#include <math.h>

// ---------------------------------------------------------------------------
// TSP policy network for MI455X (gfx1250, wave32, WMMA + async-LDS pipeline).
// ---------------------------------------------------------------------------

typedef __attribute__((ext_vector_type(16))) _Float16 v16h;
typedef __attribute__((ext_vector_type(8)))  _Float16 v8h;
typedef __attribute__((ext_vector_type(8)))  float    v8f;

#define BB 1024
#define NNK 200
#define DD 128
#define FF 512
#define HH 8
#define BN (BB * NNK)                 // 204800 rows
#define BND ((size_t)BN * DD)         // 26,214,400 floats

enum { ACT_NONE = 0, ACT_RELU = 1, ACT_GELU = 2, ACT_ELU1 = 3 };

// ---------------------------------------------------------------------------
// Async global->LDS (ASYNCcnt-tracked, CDNA5).
// ---------------------------------------------------------------------------
__device__ __forceinline__ void async_load_b128(unsigned lds_off, const float* gaddr) {
  unsigned long long ga = (unsigned long long)(size_t)gaddr;
  asm volatile("global_load_async_to_lds_b128 %0, %1, off"
               :: "v"(lds_off), "v"(ga)
               : "memory");
}
__device__ __forceinline__ void wait_async0() {
  asm volatile("s_wait_asynccnt 0x0" ::: "memory");
}

__device__ __forceinline__ void cvt8(v16h& d, int base, float4 x, float4 y) {
  d[base + 0] = (_Float16)x.x; d[base + 1] = (_Float16)x.y;
  d[base + 2] = (_Float16)x.z; d[base + 3] = (_Float16)x.w;
  d[base + 4] = (_Float16)y.x; d[base + 5] = (_Float16)y.y;
  d[base + 6] = (_Float16)y.z; d[base + 7] = (_Float16)y.w;
}

// ---------------------------------------------------------------------------
// WMMA GEMM:  C[M,Nout] = act( A[M,K] * W[Nout,K]^T + bias ) (+ residual)
// Block = 256 threads (8 waves) -> 128x64 output tile; wave = 16 rows x 64 cols
// (4 WMMAs per 32-K chunk, one A fragment reused across 4 B fragments).
// A chunks double-buffered in LDS via global_load_async_to_lds_b128 (fp32);
// B (weight) chunks staged as f16 in LDS so fragments are raw v16h loads.
// All four B fragments are materialized before the 4 back-to-back WMMAs to
// avoid the WMMA->VALU WAR hazard NOPs (ISA 7.12.1).
// ---------------------------------------------------------------------------
template <int ACT, bool RES>
__global__ void __launch_bounds__(256)
gemm_wmma_k(const float* __restrict__ A, const float* __restrict__ W,
            const float* __restrict__ bias, const float* __restrict__ Rsrc,
            float* __restrict__ C, int K, int Nout) {
  __shared__ float    Abuf[2][128][32];   // 32 KB fp32 (async target)
  __shared__ _Float16 Bh[2][64][32];      // 8 KB f16
  const int t    = threadIdx.x;           // 0..255
  const int lane = t & 31;
  const int w    = t >> 5;                // wave 0..7
  const int half = lane >> 4;
  const int l15  = lane & 15;
  const int m0 = blockIdx.x * 128;
  const int n0 = blockIdx.y * 64;

  const unsigned a_base = (unsigned)(size_t)&Abuf[0][0][0];
  const int nchunk = K >> 5;

  // ---- A tile: stream one 128x32 fp32 chunk into LDS buf (async) ----
  auto issueA = [&](int kc, int buf) {
#pragma unroll
    for (int i = 0; i < 4; ++i) {
      const int idx = i * 1024 + t * 4;
      const int m = idx >> 5, k = idx & 31;
      async_load_b128(a_base + (unsigned)(buf * 16384 + idx * 4),
                      A + (size_t)(m0 + m) * K + kc + k);
    }
  };

  // ---- B tile: 64x32 weights; 8 floats/thread held in regs, stored as f16 ----
  float4 br0, br1;
  auto loadB = [&](int kc) {
    const float* p = W + (size_t)(n0 + (t >> 2)) * K + kc + (t & 3) * 8;
    br0 = *(const float4*)p;
    br1 = *(const float4*)(p + 4);
  };
  auto storeB = [&](int buf) {
    v8h h;
    h[0] = (_Float16)br0.x; h[1] = (_Float16)br0.y;
    h[2] = (_Float16)br0.z; h[3] = (_Float16)br0.w;
    h[4] = (_Float16)br1.x; h[5] = (_Float16)br1.y;
    h[6] = (_Float16)br1.z; h[7] = (_Float16)br1.w;
    *(v8h*)&Bh[buf][t >> 2][(t & 3) * 8] = h;
  };

  v8f acc[4];
#pragma unroll
  for (int s = 0; s < 4; ++s) acc[s] = v8f{0.f,0.f,0.f,0.f,0.f,0.f,0.f,0.f};

  issueA(0, 0);
  loadB(0);
  storeB(0);
  for (int c = 0; c < nchunk; ++c) {
    const int buf = c & 1;
    wait_async0();                 // this wave's async A writes landed
    __syncthreads();               // all waves' A (async) + B (ds) writes landed
    if (c + 1 < nchunk) {
      issueA((c + 1) * 32, buf ^ 1);
      loadB((c + 1) * 32);         // global loads in flight during compute
    }

    // A fragment: contiguous k = 16*half + s (matches B fragment order)
    const float* ar = &Abuf[buf][w * 16 + l15][0];
    v16h a;
    cvt8(a, 0, *(const float4*)(ar + 16 * half),     *(const float4*)(ar + 16 * half + 4));
    cvt8(a, 8, *(const float4*)(ar + 16 * half + 8), *(const float4*)(ar + 16 * half + 12));

    // all 4 B fragments first (raw f16), then 4 hazard-free WMMAs
    v16h bf[4];
#pragma unroll
    for (int s = 0; s < 4; ++s)
      bf[s] = *(const v16h*)&Bh[buf][s * 16 + l15][16 * half];
#pragma unroll
    for (int s = 0; s < 4; ++s)
      acc[s] = __builtin_amdgcn_wmma_f32_16x16x32_f16(
          false, a, false, bf[s], (short)0, acc[s], false, false);

    if (c + 1 < nchunk) storeB(buf ^ 1);   // cvt+ds_store after WMMAs
    __syncthreads();               // everyone done reading buf before reuse
  }

  // Epilogue. C/D f32 layout: N = lane%16, M = v + 8*half.
#pragma unroll
  for (int s = 0; s < 4; ++s) {
    const int nn = n0 + s * 16 + l15;
    const float bv = bias[nn];
#pragma unroll
    for (int v = 0; v < 8; ++v) {
      const int mm = m0 + w * 16 + v + 8 * half;
      float x = acc[s][v] + bv;
      if (ACT == ACT_RELU) x = fmaxf(x, 0.f);
      else if (ACT == ACT_GELU) x = 0.5f * x * (1.f + erff(x * 0.70710678118f));
      else if (ACT == ACT_ELU1) x = (x > 0.f) ? (x + 1.f) : __expf(x);
      const size_t o = (size_t)mm * Nout + nn;
      if (RES) x += Rsrc[o];
      C[o] = x;
    }
  }
}

// ---------------------------------------------------------------------------
// LayerNorm over D=128: one wave32 per row, 4 floats per lane, shfl_xor tree.
// ---------------------------------------------------------------------------
__global__ void layernorm_k(const float* __restrict__ X,
                            const float* __restrict__ g,
                            const float* __restrict__ b,
                            float* __restrict__ Y) {
  const int wid = threadIdx.x >> 5;
  const int lane = threadIdx.x & 31;
  const size_t row = (size_t)blockIdx.x * 8 + wid;
  const float4 xv = *(const float4*)(X + row * DD + lane * 4);
  float s = xv.x + xv.y + xv.z + xv.w;
  float sq = xv.x * xv.x + xv.y * xv.y + xv.z * xv.z + xv.w * xv.w;
#pragma unroll
  for (int off = 16; off >= 1; off >>= 1) {
    s  += __shfl_xor(s, off, 32);
    sq += __shfl_xor(sq, off, 32);
  }
  const float mean = s * (1.f / DD);
  const float var = sq * (1.f / DD) - mean * mean;
  const float rstd = rsqrtf(var + 1e-5f);
  const float4 gv = *(const float4*)(g + lane * 4);
  const float4 bv = *(const float4*)(b + lane * 4);
  float4 ov;
  ov.x = (xv.x - mean) * rstd * gv.x + bv.x;
  ov.y = (xv.y - mean) * rstd * gv.y + bv.y;
  ov.z = (xv.z - mean) * rstd * gv.z + bv.z;
  ov.w = (xv.w - mean) * rstd * gv.w + bv.w;
  *(float4*)(Y + row * DD + lane * 4) = ov;
}

// ---------------------------------------------------------------------------
// Coord embedding stage 1: relu(coords @ we1^T + be1), K=2 (too small for WMMA)
// ---------------------------------------------------------------------------
__global__ void embed1_k(const float* __restrict__ coords,
                         const float* __restrict__ we1,
                         const float* __restrict__ be1,
                         float* __restrict__ T) {
  const size_t t = (size_t)blockIdx.x * blockDim.x + threadIdx.x;
  if (t >= BND) return;
  const int j = (int)(t & 127);
  const size_t m = t >> 7;
  const float v = be1[j] + coords[m * 2] * we1[j * 2] +
                  coords[m * 2 + 1] * we1[j * 2 + 1];
  T[t] = fmaxf(v, 0.f);
}

// ---------------------------------------------------------------------------
// Linear attention: kv[b,h,d,e] = sum_n k*v ; ksum[b,h,d] = sum_n k
// ---------------------------------------------------------------------------
__global__ void kv_ksum_k(const float* __restrict__ Kt,
                          const float* __restrict__ Vt,
                          float* __restrict__ KV, float* __restrict__ KS) {
  const int bh = blockIdx.x;
  const int b = bh / HH, h = bh % HH;
  const int t = threadIdx.x;           // 256
  const int d = t >> 4, e = t & 15;
  const float* kp = Kt + (size_t)b * NNK * DD + h * 16 + d;
  const float* vp = Vt + (size_t)b * NNK * DD + h * 16 + e;
  float acc = 0.f, ks = 0.f;
  for (int n = 0; n < NNK; ++n) {
    const float kk = kp[(size_t)n * DD];
    acc += kk * vp[(size_t)n * DD];
    ks += kk;
  }
  KV[(size_t)bh * 256 + t] = acc;
  if (e == 0) KS[(size_t)bh * 16 + d] = ks;
}

// out[b,n,h,e] = (q_h . kv_h[:,e]) / (q_h . ksum_h + 1e-6)
__global__ void attn_out_k(const float* __restrict__ Q,
                           const float* __restrict__ KV,
                           const float* __restrict__ KS,
                           float* __restrict__ O) {
  __shared__ float qs[DD];
  const int bn = blockIdx.x;           // b*N + n
  const int b = bn / NNK;
  const int t = threadIdx.x;           // 128
  qs[t] = Q[(size_t)bn * DD + t];
  __syncthreads();
  const int h = t >> 4, e = t & 15;
  const float* kvp = KV + ((size_t)b * HH + h) * 256;
  const float* ksp = KS + ((size_t)b * HH + h) * 16;
  const float* qh = qs + h * 16;
  float zd = 0.f, ov = 0.f;
#pragma unroll
  for (int d = 0; d < 16; ++d) {
    zd += qh[d] * ksp[d];
    ov += qh[d] * kvp[d * 16 + e];
  }
  O[(size_t)bn * DD + t] = ov / (zd + 1e-6f);
}

// ---------------------------------------------------------------------------
// Sequential GRU + pointer decoder. One block per batch row, state in LDS.
// Deterministic hash-based Gumbel-max sampling.
// ---------------------------------------------------------------------------
__global__ void decoder_k(const float* __restrict__ enc,
                          const float* __restrict__ keys,
                          const float* __restrict__ w_ih,
                          const float* __restrict__ b_ih,
                          const float* __restrict__ w_hh,
                          const float* __restrict__ b_hh,
                          const float* __restrict__ wq,
                          const float* __restrict__ bq,
                          const float* __restrict__ init_token,
                          float* __restrict__ out) {
  __shared__ float state[DD], prev[DD], gx[3 * DD], gh[3 * DD];
  __shared__ float query[DD], logits[NNK];
  __shared__ int smask[NNK];
  __shared__ int sidx;
  const int b = blockIdx.x, t = threadIdx.x;   // 256 threads
  const float* encb = enc + (size_t)b * NNK * DD;
  const float* keyb = keys + (size_t)b * NNK * DD;

  if (t < DD) {
    float s = 0.f;
    for (int n = 0; n < NNK; ++n) s += encb[(size_t)n * DD + t];
    state[t] = s * (1.f / NNK);
    prev[t] = init_token[t];
  }
  if (t < NNK) smask[t] = 1;
  __syncthreads();

  const float scale = 0.0883883476483f;  // 1/sqrt(128)
  for (int step = 0; step < NNK; ++step) {
    for (int j = t; j < 3 * DD; j += 256) {
      const float* wr = w_ih + (size_t)j * DD;
      const float* hr = w_hh + (size_t)j * DD;
      float ax = b_ih[j], ah = b_hh[j];
      for (int k = 0; k < DD; ++k) {
        ax += prev[k] * wr[k];
        ah += state[k] * hr[k];
      }
      gx[j] = ax;
      gh[j] = ah;
    }
    __syncthreads();
    if (t < DD) {
      const float r = 1.f / (1.f + __expf(-(gx[t] + gh[t])));
      const float z = 1.f / (1.f + __expf(-(gx[DD + t] + gh[DD + t])));
      const float ng = tanhf(gx[2 * DD + t] + r * gh[2 * DD + t]);
      state[t] = (1.f - z) * ng + z * state[t];
    }
    __syncthreads();
    if (t < DD) {
      const float* wr = wq + (size_t)t * DD;
      float a = bq[t];
      for (int k = 0; k < DD; ++k) a += state[k] * wr[k];
      query[t] = a;
    }
    __syncthreads();
    if (t < NNK) {
      const float* kr = keyb + (size_t)t * DD;
      float a = 0.f;
      for (int k = 0; k < DD; ++k) a += query[k] * kr[k];
      logits[t] = smask[t] ? a * scale : -INFINITY;
    }
    __syncthreads();
    if (t == 0) {
      float mx = -INFINITY;
      for (int n = 0; n < NNK; ++n) mx = fmaxf(mx, logits[n]);
      float se = 0.f;
      for (int n = 0; n < NNK; ++n) se += __expf(logits[n] - mx);
      const float lse = mx + __logf(se);
      float ent = 0.f, best = -INFINITY;
      int bi = 0;
      for (int n = 0; n < NNK; ++n) {
        if (!smask[n]) continue;
        const float lp = logits[n] - lse;
        ent -= __expf(lp) * lp;
        unsigned x = (unsigned)(b * 0x9E3779B1u) ^ (unsigned)(step * 0x85EBCA77u)
                     ^ (unsigned)(n * 0xC2B2AE3Du);
        x ^= x >> 16; x *= 0x7FEB352Du; x ^= x >> 15; x *= 0x846CA68Bu; x ^= x >> 16;
        const float u = (float)(x >> 8) * (1.f / 16777216.f) + 5.96e-8f;
        const float gum = lp - __logf(-__logf(u));
        if (gum > best) { best = gum; bi = n; }
      }
      sidx = bi;
      smask[bi] = 0;
      out[(size_t)b * NNK + step] = (float)bi;                       // tour
      out[(size_t)BN + (size_t)b * NNK + step] = logits[bi] - lse;   // logp
      out[(size_t)2 * BN + (size_t)b * NNK + step] = ent;            // entropy
    }
    __syncthreads();
    if (t < DD) prev[t] = encb[(size_t)sidx * DD + t];
    __syncthreads();
  }
}

// ---------------------------------------------------------------------------
// Host-side launch sequence
// ---------------------------------------------------------------------------
static inline void gemm_launch(int act, bool res, const float* A, const float* W,
                               const float* bias, const float* R, float* C,
                               int K, int Nout, hipStream_t s) {
  dim3 grid(BN / 128, Nout / 64);
  dim3 blk(256);
  if (res) {
    if (act == ACT_GELU) gemm_wmma_k<ACT_GELU, true><<<grid, blk, 0, s>>>(A, W, bias, R, C, K, Nout);
    else                 gemm_wmma_k<ACT_NONE, true><<<grid, blk, 0, s>>>(A, W, bias, R, C, K, Nout);
  } else {
    if (act == ACT_ELU1)      gemm_wmma_k<ACT_ELU1, false><<<grid, blk, 0, s>>>(A, W, bias, nullptr, C, K, Nout);
    else if (act == ACT_GELU) gemm_wmma_k<ACT_GELU, false><<<grid, blk, 0, s>>>(A, W, bias, nullptr, C, K, Nout);
    else                      gemm_wmma_k<ACT_NONE, false><<<grid, blk, 0, s>>>(A, W, bias, nullptr, C, K, Nout);
  }
}

extern "C" void kernel_launch(void* const* d_in, const int* in_sizes, int n_in,
                              void* d_out, int out_size, void* d_ws, size_t ws_size,
                              hipStream_t stream) {
  (void)in_sizes; (void)n_in; (void)out_size; (void)ws_size;
  // Input order: coords, then params tree-flattened (sorted keys per level):
  // be1, be2, bf, dec{b_hh,b_ih,bk,bq,init_token,w_hh,w_ih,wk,wq}, gf,
  // layers[l]{b1,b2,bf1,bf2,bk,bo,bq,bv,g1,g2,w1,w2,wk,wo,wq,wv}, we1, we2, H
  const float* coords   = (const float*)d_in[0];
  const float* be1      = (const float*)d_in[1];
  const float* be2      = (const float*)d_in[2];
  const float* bf       = (const float*)d_in[3];
  const float* dec_b_hh = (const float*)d_in[4];
  const float* dec_b_ih = (const float*)d_in[5];
  const float* dec_bk   = (const float*)d_in[6];
  const float* dec_bq   = (const float*)d_in[7];
  const float* dec_init = (const float*)d_in[8];
  const float* dec_w_hh = (const float*)d_in[9];
  const float* dec_w_ih = (const float*)d_in[10];
  const float* dec_wk   = (const float*)d_in[11];
  const float* dec_wq   = (const float*)d_in[12];
  const float* gf       = (const float*)d_in[13];
  const float* we1      = (const float*)d_in[62];
  const float* we2      = (const float*)d_in[63];

  float* X  = (float*)d_ws;       // residual stream [BN, D]
  float* T  = X + BND;            // LN out / attn mid / enc
  float* Qb = T + BND;            // q / FFN hidden (spans Qb..Eb = 4*BND)
  float* Kb = Qb + BND;
  float* Vb = Kb + BND;
  float* Eb = Vb + BND;           // 4th quarter of FFN hidden
  float* KV = Eb + BND;           // [B,H,16,16]
  float* KS = KV + (size_t)BB * HH * 256;  // [B,H,16]
  (void)Kb; (void)Vb;

  // --- embedding ---
  embed1_k<<<(unsigned)((BND + 255) / 256), 256, 0, stream>>>(coords, we1, be1, T);
  gemm_launch(ACT_NONE, false, T, we2, be2, nullptr, X, DD, DD, stream);

  // --- encoder layers ---
  for (int l = 0; l < 3; ++l) {
    const int base = 14 + 16 * l;
    const float* b1  = (const float*)d_in[base + 0];
    const float* b2  = (const float*)d_in[base + 1];
    const float* bf1 = (const float*)d_in[base + 2];
    const float* bf2 = (const float*)d_in[base + 3];
    const float* bk  = (const float*)d_in[base + 4];
    const float* bo  = (const float*)d_in[base + 5];
    const float* bq  = (const float*)d_in[base + 6];
    const float* bv  = (const float*)d_in[base + 7];
    const float* g1  = (const float*)d_in[base + 8];
    const float* g2  = (const float*)d_in[base + 9];
    const float* w1  = (const float*)d_in[base + 10];
    const float* w2  = (const float*)d_in[base + 11];
    const float* wk  = (const float*)d_in[base + 12];
    const float* wo  = (const float*)d_in[base + 13];
    const float* wq  = (const float*)d_in[base + 14];
    const float* wv  = (const float*)d_in[base + 15];

    layernorm_k<<<BN / 8, 256, 0, stream>>>(X, g1, b1, T);
    gemm_launch(ACT_ELU1, false, T, wq, bq, nullptr, Qb, DD, DD, stream);  // q = elu+1
    gemm_launch(ACT_ELU1, false, T, wk, bk, nullptr, Kb, DD, DD, stream);  // k = elu+1
    gemm_launch(ACT_NONE, false, T, wv, bv, nullptr, Vb, DD, DD, stream);  // v
    kv_ksum_k<<<BB * HH, 256, 0, stream>>>(Kb, Vb, KV, KS);
    attn_out_k<<<BN, 128, 0, stream>>>(Qb, KV, KS, T);                     // attn mid -> T
    gemm_launch(ACT_NONE, true, T, wo, bo, X, X, DD, DD, stream);          // X += proj
    layernorm_k<<<BN / 8, 256, 0, stream>>>(X, g2, b2, T);
    gemm_launch(ACT_GELU, false, T, w1, bf1, nullptr, Qb, DD, FF, stream); // hidden (4*BND)
    gemm_launch(ACT_NONE, true, Qb, w2, bf2, X, X, FF, DD, stream);        // X += ffn
  }

  // --- final LN + decoder ---
  layernorm_k<<<BN / 8, 256, 0, stream>>>(X, gf, bf, T);                   // enc -> T
  gemm_launch(ACT_NONE, false, T, dec_wk, dec_bk, nullptr, Qb, DD, DD, stream); // keys
  decoder_k<<<BB, 256, 0, stream>>>(T, Qb, dec_w_ih, dec_b_ih, dec_w_hh,
                                    dec_b_hh, dec_wq, dec_bq, dec_init,
                                    (float*)d_out);
}
